// CostVolumeReg_28020366639449
// MI455X (gfx1250) — matrix operand order repaired and checked by reading, not compiled, not run on hardware
//
#include <hip/hip_runtime.h>
#include <hip/hip_bf16.h>
#include <math.h>

typedef __attribute__((ext_vector_type(16))) _Float16 v16h;
typedef __attribute__((ext_vector_type(8)))  _Float16 v8h;
typedef __attribute__((ext_vector_type(8)))  float    v8f;

#define BN_INV 0.9999950000374997f  // 1/sqrt(1+1e-5)

static __device__ __forceinline__ float relu_f(float v) { return v > 0.f ? v : 0.f; }

static __device__ __forceinline__ v16h hz16() {
  v16h z;
#pragma unroll
  for (int j = 0; j < 16; ++j) z[j] = (_Float16)0.f;
  return z;
}
static __device__ __forceinline__ v8h hz8() {
  v8h z;
#pragma unroll
  for (int j = 0; j < 8; ++j) z[j] = (_Float16)0.f;
  return z;
}
static __device__ __forceinline__ v8f fz8() {
  v8f z = {0.f, 0.f, 0.f, 0.f, 0.f, 0.f, 0.f, 0.f};
  return z;
}

// ---------------------------------------------------------------------------
// Compile-time B-fragment LDS offset tables. K order is tap-major (k =
// tap*CIN + ci), so a lane's 16 consecutive K values are one contiguous
// 32B-aligned channel run at a fixed tap. The tap->offset mapping depends
// only on (chunk, lane-half) and template constants, so it folds to literals
// once the chunk loop is unrolled: per chunk = one cndmask + the LDS load.
// ---------------------------------------------------------------------------
template<int CIN, int TAPS, int WT, int NCHUNK, int KB>
struct OffTab3 {  // offset = ((kd*3+kh)*WT + kw)*CIN + ci0 ; -1 if tap OOB
  int v[NCHUNK];
  constexpr OffTab3() : v() {
    for (int c = 0; c < NCHUNK; ++c) {
      int ks = c * 32 + KB;
      int tap = ks / CIN;
      if (tap >= TAPS) { v[c] = -1; continue; }
      int ci0 = ks - tap * CIN;
      int kd = tap / 9, t9 = tap - kd * 9, kh = t9 / 3, kw = t9 - kh * 3;
      v[c] = ((kd * 3 + kh) * WT + kw) * CIN + ci0;
    }
  }
};

template<int CIN, int WT, int NCHUNK, int KB>
struct OffTab2 {  // 2D conv: offset = (kh*WT + kw)*CIN + ci0 (tap < 9 always)
  int v[NCHUNK];
  constexpr OffTab2() : v() {
    for (int c = 0; c < NCHUNK; ++c) {
      int ks = c * 32 + KB;
      int tap = ks / CIN;
      int ci0 = ks - tap * CIN;
      int kh = tap / 3, kw = tap - kh * 3;
      v[c] = (kh * WT + kw) * CIN + ci0;
    }
  }
};

template<int CIN, int IWT, int NCHUNK, int KB>
struct OffTabT {  // convT: addr = base[c] + ((n+1-kw)>>1)*CIN, valid if even
  int base[NCHUNK];
  int kwv[NCHUNK];
  constexpr OffTabT() : base(), kwv() {
    for (int c = 0; c < NCHUNK; ++c) {
      int ks = c * 32 + KB;
      int tap = ks / CIN;
      int ci0 = ks - tap * CIN;
      int td = tap >> 3, th = (tap >> 2) & 1, kw = tap & 3;
      base[c] = (((1 - td) * 2 + (1 - th)) * IWT + 1) * CIN + ci0;  // +1 = lx bias
      kwv[c] = kw;
    }
  }
};

// ---------------------------------------------------------------------------
// Weight packing into WMMA A-fragment lane layout (tap-major K):
//   lane<16 : M=lane,    half j -> K = j
//   lane>=16: M=lane-16, half j -> K = j+16
// Packed: [MT][NCHUNK][32 lanes][16 halves]; each lane reads 32B contiguous.
// ---------------------------------------------------------------------------
template<int CIN, int TAPS, int COUT>
__global__ void pack_wk(const float* __restrict__ w, _Float16* __restrict__ apack) {
  constexpr int K = CIN * TAPS;
  constexpr int NCHUNK = (K + 31) / 32;
  constexpr int MT = (COUT + 15) / 16;
  const int total = MT * NCHUNK * 512;
  for (int t = blockIdx.x * blockDim.x + threadIdx.x; t < total; t += gridDim.x * blockDim.x) {
    int j = t & 15;
    int lane = (t >> 4) & 31;
    int rest = t >> 9;
    int chunk = rest % NCHUNK;
    int mt = rest / NCHUNK;
    int m = mt * 16 + (lane & 15);
    int klocal = (lane < 16) ? j : j + 16;
    int k = chunk * 32 + klocal;
    _Float16 v = (_Float16)0.f;
    if (m < COUT && k < K) {
      int tap = k / CIN;
      int ci = k - tap * CIN;
      v = (_Float16)w[((size_t)m * CIN + ci) * TAPS + tap];  // w: [COUT][CIN][TAPS]
    }
    apack[t] = v;
  }
}

// Transposed-conv weights [CIN][COUT][4][4][4]; k = tap*CIN + i with
// tap = td*8 + th*4 + kw ; kd = pz + 2*td, kh = py + 2*th.
// Packed per output (z,y) parity: [par=pz*2+py][NCHUNK][32][16].
template<int CIN, int COUT>
__global__ void pack_wT(const float* __restrict__ w, _Float16* __restrict__ apack) {
  constexpr int K = CIN * 16;
  constexpr int NCHUNK = K / 32;
  const int total = 4 * NCHUNK * 512;
  for (int t = blockIdx.x * blockDim.x + threadIdx.x; t < total; t += gridDim.x * blockDim.x) {
    int j = t & 15;
    int lane = (t >> 4) & 31;
    int rest = t >> 9;
    int chunk = rest % NCHUNK;
    int par = rest / NCHUNK;
    int pz = par >> 1, py = par & 1;
    int m = lane & 15;
    int klocal = (lane < 16) ? j : j + 16;
    int k = chunk * 32 + klocal;
    int tap = k / CIN;
    int i = k - tap * CIN;
    int td = tap >> 3, th = (tap >> 2) & 1, kw = tap & 3;
    int kd = pz + 2 * td, kh = py + 2 * th;
    _Float16 v = (_Float16)0.f;
    if (m < COUT) v = (_Float16)w[(((size_t)i * COUT + m) * 4 + kd) * 16 + kh * 4 + kw];
    apack[t] = v;
  }
}

// ---------------------------------------------------------------------------
// 3x3x3 conv, pad 1, stride 1|2, implicit GEMM on v_wmma_f32_16x16x32_f16.
// One wave per (b, d_out, h_out, 16-wide w tile). Halo staged to LDS with
// channel-innermost layout; B fragment = one contiguous v16h LDS load at a
// compile-time per-chunk offset. Fused BN scale/bias (+opt bias) + ReLU.
// ---------------------------------------------------------------------------
template<int CIN, int COUT, int STRIDE, bool RELU, bool OUTF32>
__global__ __launch_bounds__(32) void conv3d3_wmma(
    const _Float16* __restrict__ x, const _Float16* __restrict__ apack,
    const float* __restrict__ bng, const float* __restrict__ bnb,
    const float* __restrict__ cbias, void* __restrict__ yout,
    int B, int Din, int Hin, int Win, int Do, int Ho, int Wo) {
  constexpr int TAPS = 27;
  constexpr int K = CIN * TAPS;
  constexpr int NCHUNK = (K + 31) / 32;
  constexpr int MT = (COUT + 15) / 16;
  constexpr int WT = 15 * STRIDE + 3;
  __shared__ _Float16 xt[9 * WT * CIN];  // [(kd*3+kh)*WT + wx][ci]

  const int lane = threadIdx.x;
  const int wtiles = (Wo + 15) >> 4;
  int tile = blockIdx.x;
  const int wt = tile % wtiles; tile /= wtiles;
  const int ho = tile % Ho;     tile /= Ho;
  const int dq = tile % Do;     tile /= Do;
  const int b  = tile;

  const int w0 = wt * 16 * STRIDE - 1;
  const int h0 = ho * STRIDE - 1;
  const int d0 = dq * STRIDE - 1;
  const size_t HWi = (size_t)Hin * Win;
  const size_t DHWi = (size_t)Din * HWi;

  // stage halo tile: global reads coalesced along wx; store channel-innermost
  for (int t = lane; t < CIN * 9 * WT; t += 32) {
    int wx = t % WT;
    int r = t / WT;           // (ci*3+kd)*3+kh
    int kh = r % 3; r /= 3;
    int kd = r % 3;
    int ci = r / 3;
    int dz = d0 + kd, hy = h0 + kh, wz = w0 + wx;
    _Float16 v = (_Float16)0.f;
    if ((unsigned)dz < (unsigned)Din && (unsigned)hy < (unsigned)Hin && (unsigned)wz < (unsigned)Win)
      v = x[(size_t)(b * CIN + ci) * DHWi + (size_t)dz * HWi + (size_t)hy * Win + wz];
    xt[(size_t)((kd * 3 + kh) * WT + wx) * CIN + ci] = v;
  }
  __syncthreads();

  v8f acc[MT];
#pragma unroll
  for (int mt = 0; mt < MT; ++mt) acc[mt] = fz8();

  const int n = lane & 15;
  const int hi = lane >> 4;
  const _Float16* xb = xt + (size_t)(n * STRIDE) * CIN;

#pragma unroll
  for (int c = 0; c < NCHUNK; ++c) {
    v16h bf;
    if constexpr (CIN >= 16) {
      constexpr OffTab3<CIN, TAPS, WT, NCHUNK, 0>  tA{};
      constexpr OffTab3<CIN, TAPS, WT, NCHUNK, 16> tB{};
      const int off = hi ? tB.v[c] : tA.v[c];
      if (off >= 0) bf = *(const v16h*)(xb + off);
      else          bf = hz16();
    } else {  // CIN == 8: 16 K values span two taps (8 channels each)
      constexpr OffTab3<8, TAPS, WT, NCHUNK, 0>  tA0{};
      constexpr OffTab3<8, TAPS, WT, NCHUNK, 8>  tA1{};
      constexpr OffTab3<8, TAPS, WT, NCHUNK, 16> tB0{};
      constexpr OffTab3<8, TAPS, WT, NCHUNK, 24> tB1{};
      const int offA = hi ? tB0.v[c] : tA0.v[c];
      const int offB = hi ? tB1.v[c] : tA1.v[c];
      v8h lo = (offA >= 0) ? *(const v8h*)(xb + offA) : hz8();
      v8h hi8 = (offB >= 0) ? *(const v8h*)(xb + offB) : hz8();
#pragma unroll
      for (int j = 0; j < 8; ++j) { bf[j] = lo[j]; bf[j + 8] = hi8[j]; }
    }
#pragma unroll
    for (int mt = 0; mt < MT; ++mt) {
      const v16h af = *(const v16h*)(apack + ((size_t)(mt * NCHUNK + c) * 32 + lane) * 16);
      acc[mt] = __builtin_amdgcn_wmma_f32_16x16x32_f16(false, af, false, bf, (short)0,
                                                       acc[mt], false, false);
    }
  }

  const int wo = wt * 16 + n;
  if (wo < Wo) {
    const size_t oHW = (size_t)Ho * Wo;
    const int mbase = hi ? 8 : 0;  // C/D layout: VGPR r -> M=r (+8 hi lanes)
#pragma unroll
    for (int mt = 0; mt < MT; ++mt) {
#pragma unroll
      for (int r = 0; r < 8; ++r) {
        int co = mt * 16 + mbase + r;
        if (co < COUT) {
          float vv = acc[mt][r];
          float scale = bng ? bng[co] * BN_INV : 1.f;
          float bias = (bnb ? bnb[co] : 0.f) + (cbias ? cbias[co] * scale : 0.f);
          vv = vv * scale + bias;
          if (RELU) vv = relu_f(vv);
          size_t oi = ((size_t)(b * COUT + co) * Do + dq) * oHW + (size_t)ho * Wo + wo;
          if (OUTF32) ((float*)yout)[oi] = vv;
          else        ((_Float16*)yout)[oi] = (_Float16)vv;
        }
      }
    }
  }
}

// ---------------------------------------------------------------------------
// 3x3 2D conv, stride 2, pad 1 (gfp1/gfp2), tap-major K, bias+BN+ReLU.
// ---------------------------------------------------------------------------
template<int CIN, int COUT>
__global__ __launch_bounds__(32) void conv2d3s2_wmma(
    const _Float16* __restrict__ x, const _Float16* __restrict__ apack,
    const float* __restrict__ cbias, const float* __restrict__ bng,
    const float* __restrict__ bnb, _Float16* __restrict__ y,
    int B, int Hin, int Win, int Ho, int Wo) {
  constexpr int K = CIN * 9;           // 576 / 864 : exact multiples of 32
  constexpr int NCHUNK = K / 32;
  constexpr int MT = (COUT + 15) / 16;
  constexpr int WT = 33;
  __shared__ _Float16 xt[3 * WT * CIN];  // [kh*WT + wx][ci]

  const int lane = threadIdx.x;
  const int wtiles = (Wo + 15) >> 4;
  int tile = blockIdx.x;
  const int wt = tile % wtiles; tile /= wtiles;
  const int ho = tile % Ho;     tile /= Ho;
  const int b  = tile;
  const int w0 = wt * 32 - 1;
  const int h0 = ho * 2 - 1;
  const size_t HWi = (size_t)Hin * Win;

  for (int t = lane; t < CIN * 3 * WT; t += 32) {
    int wx = t % WT;
    int r = t / WT;
    int kh = r % 3;
    int ci = r / 3;
    int hy = h0 + kh, wz = w0 + wx;
    _Float16 v = (_Float16)0.f;
    if ((unsigned)hy < (unsigned)Hin && (unsigned)wz < (unsigned)Win)
      v = x[(size_t)(b * CIN + ci) * HWi + (size_t)hy * Win + wz];
    xt[(size_t)(kh * WT + wx) * CIN + ci] = v;
  }
  __syncthreads();

  v8f acc[MT];
#pragma unroll
  for (int mt = 0; mt < MT; ++mt) acc[mt] = fz8();

  const int n = lane & 15;
  const int hi = lane >> 4;
  const _Float16* xb = xt + (size_t)(n * 2) * CIN;

#pragma unroll
  for (int c = 0; c < NCHUNK; ++c) {
    constexpr OffTab2<CIN, WT, NCHUNK, 0>  tA{};
    constexpr OffTab2<CIN, WT, NCHUNK, 16> tB{};
    const int off = hi ? tB.v[c] : tA.v[c];
    const v16h bf = *(const v16h*)(xb + off);
#pragma unroll
    for (int mt = 0; mt < MT; ++mt) {
      const v16h af = *(const v16h*)(apack + ((size_t)(mt * NCHUNK + c) * 32 + lane) * 16);
      acc[mt] = __builtin_amdgcn_wmma_f32_16x16x32_f16(false, af, false, bf, (short)0,
                                                       acc[mt], false, false);
    }
  }

  const int wo = wt * 16 + n;
  if (wo < Wo) {
    const int mbase = hi ? 8 : 0;
#pragma unroll
    for (int mt = 0; mt < MT; ++mt) {
#pragma unroll
      for (int r = 0; r < 8; ++r) {
        int co = mt * 16 + mbase + r;
        if (co < COUT) {
          float scale = bng[co] * BN_INV;
          float vv = acc[mt][r] * scale + (bnb[co] + cbias[co] * scale);
          y[((size_t)(b * COUT + co) * Ho + ho) * Wo + wo] = (_Float16)relu_f(vv);
        }
      }
    }
  }
}

// ---------------------------------------------------------------------------
// ConvTranspose3d k=4 s=2 p=1 via WMMA. K = 16 taps * CIN, tap-major;
// output (z,y) parity selects one of 4 pre-packed A variants. bias+BN+ReLU.
// B address: base[c] (compile-time) + ((n+1-kw)>>1)*CIN, valid iff even.
// ---------------------------------------------------------------------------
template<int CIN, int COUT>
__global__ __launch_bounds__(32) void convT3d_wmma(
    const _Float16* __restrict__ x, const _Float16* __restrict__ apackT,
    const float* __restrict__ cbias, const float* __restrict__ bng,
    const float* __restrict__ bnb, _Float16* __restrict__ y,
    int B, int Din, int Hin, int Win) {
  constexpr int K = CIN * 16;
  constexpr int NCHUNK = K / 32;
  constexpr int IWT = 10;
  __shared__ _Float16 xt[4 * IWT * CIN];  // [(zl*2+yl)*IWT + lx][ci]

  const int Do = 2 * Din, Ho = 2 * Hin, Wo = 2 * Win;
  const int lane = threadIdx.x;
  const int wtiles = Wo >> 4;
  int tile = blockIdx.x;
  const int wt = tile % wtiles; tile /= wtiles;
  const int yy = tile % Ho;     tile /= Ho;
  const int zz = tile % Do;     tile /= Do;
  const int b  = tile;

  const int pz = (zz + 1) & 1, py = (yy + 1) & 1;
  const int par = pz * 2 + py;
  const int izb = (zz - 1 - pz) / 2;
  const int iyb = (yy - 1 - py) / 2;
  const int x0 = wt * 16;                 // even
  const int ixb = (x0 >> 1) - 1;
  const size_t HWi = (size_t)Hin * Win;
  const size_t DHWi = (size_t)Din * HWi;

  for (int t = lane; t < CIN * 4 * IWT; t += 32) {
    int lx = t % IWT;
    int r = t / IWT;
    int yl = r & 1;
    int zl = (r >> 1) & 1;
    int ci = r >> 2;
    int iz = izb + zl, iy = iyb + yl, ix = ixb + lx;
    _Float16 v = (_Float16)0.f;
    if ((unsigned)iz < (unsigned)Din && (unsigned)iy < (unsigned)Hin && (unsigned)ix < (unsigned)Win)
      v = x[(size_t)(b * CIN + ci) * DHWi + (size_t)iz * HWi + (size_t)iy * Win + ix];
    xt[(size_t)((zl * 2 + yl) * IWT + lx) * CIN + ci] = v;
  }
  __syncthreads();

  v8f acc = fz8();
  const int n = lane & 15;
  const int hi = lane >> 4;
  const int nn = n + 1;

#pragma unroll
  for (int c = 0; c < NCHUNK; ++c) {
    constexpr OffTabT<CIN, IWT, NCHUNK, 0>  tA{};
    constexpr OffTabT<CIN, IWT, NCHUNK, 16> tB{};
    const int base = hi ? tB.base[c] : tA.base[c];
    const int kw   = hi ? tB.kwv[c]  : tA.kwv[c];
    const int d = nn - kw;
    v16h bf = hz16();
    if ((d & 1) == 0) bf = *(const v16h*)(xt + base + (d >> 1) * CIN);
    const v16h af = *(const v16h*)(apackT + ((size_t)(par * NCHUNK + c) * 32 + lane) * 16);
    acc = __builtin_amdgcn_wmma_f32_16x16x32_f16(false, af, false, bf, (short)0, acc, false, false);
  }

  const int wo = x0 + n;
  const int mbase = hi ? 8 : 0;
  const size_t oHW = (size_t)Ho * Wo;
#pragma unroll
  for (int r = 0; r < 8; ++r) {
    int co = mbase + r;
    if (co < COUT && wo < Wo) {
      float scale = bng[co] * BN_INV;
      float vv = acc[r] * scale + (bnb[co] + cbias[co] * scale);
      y[((size_t)(b * COUT + co) * Do + zz) * oHW + (size_t)yy * Wo + wo] = (_Float16)relu_f(vv);
    }
  }
}

// ---------------------------------------------------------------------------
// VALU kernels (bandwidth-bound / tiny paths).
// ---------------------------------------------------------------------------
__global__ void f32_to_f16_kernel(const float* __restrict__ x, _Float16* __restrict__ y, size_t n) {
  size_t i = (size_t)blockIdx.x * blockDim.x + threadIdx.x;
  if (i < n) y[i] = (_Float16)x[i];
}

// group-wise correlation: vol[b,g,d,h,w] = mean_c ref[b,g*8+c,h,w]*tgt[b,g*8+c,h,w-d]
__global__ void gwc_kernel(const float* __restrict__ ref, const float* __restrict__ tgt,
                           _Float16* __restrict__ vol, size_t total) {
  size_t idx = (size_t)blockIdx.x * blockDim.x + threadIdx.x;
  if (idx >= total) return;
  const int W = 320, H = 96, Dd = 48, G = 8;
  int w = (int)(idx % W); size_t r = idx / W;
  int h = (int)(r % H);   r /= H;
  int d = (int)(r % Dd);  r /= Dd;
  int g = (int)(r % G);
  int b = (int)(r / G);
  float s = 0.f;
  if (w >= d) {
    const size_t HW = (size_t)H * W;
    const float* rp = ref + (size_t)(b * 64 + g * 8) * HW + (size_t)h * W + w;
    const float* tp = tgt + (size_t)(b * 64 + g * 8) * HW + (size_t)h * W + (w - d);
#pragma unroll
    for (int c = 0; c < 8; ++c) s += rp[(size_t)c * HW] * tp[(size_t)c * HW];
    s *= 0.125f;
  }
  vol[idx] = (_Float16)s;
}

// 1x1 2D conv; MODE 0: conv+bias -> BN -> ReLU ; MODE 1: conv+bias -> sigmoid
template<int CIN, int COUT, int MODE>
__global__ void conv1x1_2d(const _Float16* __restrict__ x, const float* __restrict__ w,
                           const float* __restrict__ cbias, const float* __restrict__ bng,
                           const float* __restrict__ bnb, _Float16* __restrict__ y,
                           int B, size_t HW) {
  size_t idx = (size_t)blockIdx.x * blockDim.x + threadIdx.x;
  size_t total = (size_t)B * COUT * HW;
  if (idx >= total) return;
  size_t hw = idx % HW;
  size_t r = idx / HW;
  int co = (int)(r % COUT);
  int b = (int)(r / COUT);
  const _Float16* xp = x + (size_t)b * CIN * HW + hw;
  const float* wp = w + (size_t)co * CIN;
  float s = cbias[co];
#pragma unroll 8
  for (int ci = 0; ci < CIN; ++ci) s += (float)xp[(size_t)ci * HW] * wp[ci];
  if (MODE == 0) { s = relu_f(s * bng[co] * BN_INV + bnb[co]); }
  else           { s = 1.f / (1.f + __expf(-s)); }
  y[idx] = (_Float16)s;
}

// feature-attention gating: cv[b,c,d,hw] *= gate[b,d,hw]
__global__ void gate_mul_kernel(_Float16* __restrict__ cv, const _Float16* __restrict__ gate,
                                int C, int Dd, size_t HW, size_t total) {
  size_t idx = (size_t)blockIdx.x * blockDim.x + threadIdx.x;
  if (idx >= total) return;
  size_t hw = idx % HW;
  size_t r = idx / HW;
  int d = (int)(r % Dd); r /= Dd;
  r /= C;
  int b = (int)r;
  float g = (float)gate[((size_t)b * Dd + d) * HW + hw];
  cv[idx] = (_Float16)((float)cv[idx] * g);
}

// 1x1x1 conv over concat([src0, src1]) -> CH channels, in place on src0.
template<int CH>
__global__ void fuse1x1_3d(_Float16* __restrict__ src0, const _Float16* __restrict__ src1,
                           const float* __restrict__ w, const float* __restrict__ cbias,
                           const float* __restrict__ bng, const float* __restrict__ bnb,
                           int B, size_t VOX) {
  size_t idx = (size_t)blockIdx.x * blockDim.x + threadIdx.x;
  if (idx >= (size_t)B * VOX) return;
  size_t v = idx % VOX;
  int b = (int)(idx / VOX);
  _Float16* p0 = src0 + (size_t)b * CH * VOX + v;
  const _Float16* p1 = src1 + (size_t)b * CH * VOX + v;
  float in0[CH], in1[CH], out[CH];
#pragma unroll
  for (int c = 0; c < CH; ++c) { in0[c] = (float)p0[(size_t)c * VOX]; in1[c] = (float)p1[(size_t)c * VOX]; }
#pragma unroll
  for (int co = 0; co < CH; ++co) {
    const float* wp = w + (size_t)co * 2 * CH;
    float s = cbias[co];
#pragma unroll
    for (int ci = 0; ci < CH; ++ci) s += wp[ci] * in0[ci];
#pragma unroll
    for (int ci = 0; ci < CH; ++ci) s += wp[CH + ci] * in1[ci];
    out[co] = relu_f(s * bng[co] * BN_INV + bnb[co]);
  }
#pragma unroll
  for (int c = 0; c < CH; ++c) p0[(size_t)c * VOX] = (_Float16)out[c];
}

__global__ void softmax48_kernel(const float* __restrict__ logits, float* __restrict__ out,
                                 int B, size_t HW) {
  size_t idx = (size_t)blockIdx.x * blockDim.x + threadIdx.x;
  if (idx >= (size_t)B * HW) return;
  size_t hw = idx % HW;
  int b = (int)(idx / HW);
  const float* p = logits + (size_t)b * 48 * HW + hw;
  float e[48];
  float mx = -1e30f;
#pragma unroll
  for (int d = 0; d < 48; ++d) { e[d] = p[(size_t)d * HW]; mx = fmaxf(mx, e[d]); }
  float se = 0.f;
#pragma unroll
  for (int d = 0; d < 48; ++d) { float t = __expf(e[d] - mx); e[d] = t; se += t; }
  float inv = 1.f / se;
  float* q = out + (size_t)b * 48 * HW + hw;
#pragma unroll
  for (int d = 0; d < 48; ++d) q[(size_t)d * HW] = e[d] * inv;
}

// ---------------------------------------------------------------------------
// Host orchestration
// ---------------------------------------------------------------------------
static inline int cdiv_i(size_t a, size_t b) { return (int)((a + b - 1) / b); }

extern "C" void kernel_launch(void* const* d_in, const int* in_sizes, int n_in,
                              void* d_out, int out_size, void* d_ws, size_t ws_size,
                              hipStream_t stream) {
  (void)in_sizes; (void)n_in; (void)out_size; (void)ws_size;
  const float* left  = (const float*)d_in[0];
  const float* right = (const float*)d_in[1];
#define PF(i) ((const float*)d_in[(i)])
  const float* stem_w = PF(2),  *stem_g = PF(3),  *stem_b = PF(4);
  const float* fa0_w1 = PF(5),  *fa0_b1 = PF(6),  *fa0_g = PF(7),  *fa0_bb = PF(8),  *fa0_w2 = PF(9),  *fa0_b2 = PF(10);
  const float* fa1_w1 = PF(11), *fa1_b1 = PF(12), *fa1_g = PF(13), *fa1_bb = PF(14), *fa1_w2 = PF(15), *fa1_b2 = PF(16);
  const float* fa2_w1 = PF(17), *fa2_b1 = PF(18), *fa2_g = PF(19), *fa2_bb = PF(20), *fa2_w2 = PF(21), *fa2_b2 = PF(22);
  const float* oa_w1  = PF(23), *oa_b1  = PF(24), *oa_g  = PF(25), *oa_bb  = PF(26), *oa_w2  = PF(27), *oa_b2  = PF(28);
  const float* gfp1_w = PF(29), *gfp1_b = PF(30), *gfp1_g = PF(31), *gfp1_bb = PF(32);
  const float* gfp2_w = PF(33), *gfp2_b = PF(34), *gfp2_g = PF(35), *gfp2_bb = PF(36);
  const float* c1a_w = PF(37), *c1a_g = PF(38), *c1a_b = PF(39);
  const float* c1b_w = PF(40), *c1b_g = PF(41), *c1b_b = PF(42);
  const float* c2a_w = PF(43), *c2a_g = PF(44), *c2a_b = PF(45);
  const float* c2b_w = PF(46), *c2b_g = PF(47), *c2b_b = PF(48);
  const float* c2u_w = PF(49), *c2u_b = PF(50), *c2u_g = PF(51), *c2u_bb = PF(52);
  const float* a0a_w = PF(53), *a0a_b = PF(54), *a0a_g = PF(55), *a0a_bb = PF(56);
  const float* a0b_w = PF(57), *a0b_b = PF(58), *a0b_g = PF(59), *a0b_bb = PF(60);
  const float* c1u_w = PF(61), *c1u_b = PF(62), *c1u_g = PF(63), *c1u_bb = PF(64);
  const float* a1a_w = PF(65), *a1a_b = PF(66), *a1a_g = PF(67), *a1a_bb = PF(68);
  const float* a1b_w = PF(69), *a1b_b = PF(70), *a1b_g = PF(71), *a1b_bb = PF(72);
  const float* out_w = PF(73);
#undef PF

  const int B = 4;
  const size_t HW0 = 96 * 320, HW1 = 48 * 160, HW2 = 24 * 80;
  const size_t VOL0 = (size_t)48 * HW0;
  const size_t VOL1 = (size_t)24 * HW1;
  const size_t VOL2 = (size_t)12 * HW2;

  char* wsb = (char*)d_ws;
  size_t off = 0;
  auto take = [&](size_t elems, size_t esz) -> void* {
    void* p = wsb + off;
    off += ((elems * esz + 255) & ~(size_t)255);
    return p;
  };
  _Float16* BUF0  = (_Float16*)take((size_t)B * 8 * VOL0, 2);   // vol -> c1u/a1a(in-place)
  _Float16* BUF1  = (_Float16*)take((size_t)B * 8 * VOL0, 2);   // cost -> a1b out/final
  _Float16* featL = (_Float16*)take((size_t)B * 64 * HW0, 2);
  _Float16* hid0  = (_Float16*)take((size_t)B * 32 * HW0, 2);
  _Float16* gate0 = (_Float16*)take((size_t)B * 48 * HW0, 2);
  _Float16* feat1 = (_Float16*)take((size_t)B * 96 * HW1, 2);
  _Float16* hid1  = (_Float16*)take((size_t)B * 48 * HW1, 2);
  _Float16* gate1 = (_Float16*)take((size_t)B * 24 * HW1, 2);
  _Float16* feat2 = (_Float16*)take((size_t)B * 128 * HW2, 2);
  _Float16* hid2  = (_Float16*)take((size_t)B * 64 * HW2, 2);
  _Float16* gate2 = (_Float16*)take((size_t)B * 12 * HW2, 2);
  _Float16* c1A   = (_Float16*)take((size_t)B * 16 * VOL1, 2);  // c1a out -> a0b out
  _Float16* c1B   = (_Float16*)take((size_t)B * 16 * VOL1, 2);
  _Float16* c2A   = (_Float16*)take((size_t)B * 32 * VOL2, 2);
  _Float16* c2B   = (_Float16*)take((size_t)B * 32 * VOL2, 2);
  _Float16* c2uB  = (_Float16*)take((size_t)B * 16 * VOL1, 2);  // c2u out -> a0a in-place
  float*    logits= (float*)   take((size_t)B * VOL0, 4);
  _Float16* pk_stem = (_Float16*)take(1 * 7 * 512, 2);
  _Float16* pk_c1a  = (_Float16*)take(1 * 7 * 512, 2);
  _Float16* pk_c1b  = (_Float16*)take(1 * 14 * 512, 2);
  _Float16* pk_c2a  = (_Float16*)take(2 * 14 * 512, 2);
  _Float16* pk_c2b  = (_Float16*)take(2 * 27 * 512, 2);
  _Float16* pk_a0b  = (_Float16*)take(1 * 14 * 512, 2);
  _Float16* pk_a1b  = (_Float16*)take(1 * 7 * 512, 2);
  _Float16* pk_out  = (_Float16*)take(1 * 7 * 512, 2);
  _Float16* pk_g1   = (_Float16*)take(6 * 18 * 512, 2);
  _Float16* pk_g2   = (_Float16*)take(8 * 27 * 512, 2);
  _Float16* pk_c2u  = (_Float16*)take(4 * 16 * 512, 2);
  _Float16* pk_c1u  = (_Float16*)take(4 * 8 * 512, 2);

  // ---- weight packing (tap-major K order) ----
  pack_wk<8, 27, 8>   <<<14, 256, 0, stream>>>(stem_w, pk_stem);
  pack_wk<8, 27, 16>  <<<14, 256, 0, stream>>>(c1a_w, pk_c1a);
  pack_wk<16, 27, 16> <<<28, 256, 0, stream>>>(c1b_w, pk_c1b);
  pack_wk<16, 27, 32> <<<56, 256, 0, stream>>>(c2a_w, pk_c2a);
  pack_wk<32, 27, 32> <<<108, 256, 0, stream>>>(c2b_w, pk_c2b);
  pack_wk<16, 27, 16> <<<28, 256, 0, stream>>>(a0b_w, pk_a0b);
  pack_wk<8, 27, 8>   <<<14, 256, 0, stream>>>(a1b_w, pk_a1b);
  pack_wk<8, 27, 1>   <<<14, 256, 0, stream>>>(out_w, pk_out);
  pack_wk<64, 9, 96>  <<<216, 256, 0, stream>>>(gfp1_w, pk_g1);
  pack_wk<96, 9, 128> <<<432, 256, 0, stream>>>(gfp2_w, pk_g2);
  pack_wT<32, 16>     <<<128, 256, 0, stream>>>(c2u_w, pk_c2u);
  pack_wT<16, 8>      <<<64, 256, 0, stream>>>(c1u_w, pk_c1u);

  // ---- feature prep + correlation volume ----
  f32_to_f16_kernel<<<cdiv_i((size_t)B * 64 * HW0, 256), 256, 0, stream>>>(left, featL, (size_t)B * 64 * HW0);
  gwc_kernel<<<cdiv_i((size_t)B * 8 * VOL0, 256), 256, 0, stream>>>(left, right, BUF0, (size_t)B * 8 * VOL0);

  // ---- stem ----
  conv3d3_wmma<8, 8, 1, true, false><<<B * 48 * 96 * 20, 32, 0, stream>>>(
      BUF0, pk_stem, stem_g, stem_b, nullptr, BUF1, B, 48, 96, 320, 48, 96, 320);

  // ---- fa0 gate ----
  conv1x1_2d<64, 32, 0><<<cdiv_i((size_t)B * 32 * HW0, 256), 256, 0, stream>>>(featL, fa0_w1, fa0_b1, fa0_g, fa0_bb, hid0, B, HW0);
  conv1x1_2d<32, 48, 1><<<cdiv_i((size_t)B * 48 * HW0, 256), 256, 0, stream>>>(hid0, fa0_w2, fa0_b2, nullptr, nullptr, gate0, B, HW0);
  gate_mul_kernel<<<cdiv_i((size_t)B * 8 * VOL0, 256), 256, 0, stream>>>(BUF1, gate0, 8, 48, HW0, (size_t)B * 8 * VOL0);

  // ---- gfp1 ----
  conv2d3s2_wmma<64, 96><<<B * 48 * 10, 32, 0, stream>>>(featL, pk_g1, gfp1_b, gfp1_g, gfp1_bb, feat1, B, 96, 320, 48, 160);

  // ---- c1a / c1b ----
  conv3d3_wmma<8, 16, 2, true, false><<<B * 24 * 48 * 10, 32, 0, stream>>>(
      BUF1, pk_c1a, c1a_g, c1a_b, nullptr, c1A, B, 48, 96, 320, 24, 48, 160);
  conv3d3_wmma<16, 16, 1, true, false><<<B * 24 * 48 * 10, 32, 0, stream>>>(
      c1A, pk_c1b, c1b_g, c1b_b, nullptr, c1B, B, 24, 48, 160, 24, 48, 160);

  // ---- fa1 gate ----
  conv1x1_2d<96, 48, 0><<<cdiv_i((size_t)B * 48 * HW1, 256), 256, 0, stream>>>(feat1, fa1_w1, fa1_b1, fa1_g, fa1_bb, hid1, B, HW1);
  conv1x1_2d<48, 24, 1><<<cdiv_i((size_t)B * 24 * HW1, 256), 256, 0, stream>>>(hid1, fa1_w2, fa1_b2, nullptr, nullptr, gate1, B, HW1);
  gate_mul_kernel<<<cdiv_i((size_t)B * 16 * VOL1, 256), 256, 0, stream>>>(c1B, gate1, 16, 24, HW1, (size_t)B * 16 * VOL1);

  // ---- gfp2 ----
  conv2d3s2_wmma<96, 128><<<B * 24 * 5, 32, 0, stream>>>(feat1, pk_g2, gfp2_b, gfp2_g, gfp2_bb, feat2, B, 48, 160, 24, 80);

  // ---- c2a / c2b ----
  conv3d3_wmma<16, 32, 2, true, false><<<B * 12 * 24 * 5, 32, 0, stream>>>(
      c1B, pk_c2a, c2a_g, c2a_b, nullptr, c2A, B, 24, 48, 160, 12, 24, 80);
  conv3d3_wmma<32, 32, 1, true, false><<<B * 12 * 24 * 5, 32, 0, stream>>>(
      c2A, pk_c2b, c2b_g, c2b_b, nullptr, c2B, B, 12, 24, 80, 12, 24, 80);

  // ---- fa2 gate ----
  conv1x1_2d<128, 64, 0><<<cdiv_i((size_t)B * 64 * HW2, 256), 256, 0, stream>>>(feat2, fa2_w1, fa2_b1, fa2_g, fa2_bb, hid2, B, HW2);
  conv1x1_2d<64, 12, 1><<<cdiv_i((size_t)B * 12 * HW2, 256), 256, 0, stream>>>(hid2, fa2_w2, fa2_b2, nullptr, nullptr, gate2, B, HW2);
  gate_mul_kernel<<<cdiv_i((size_t)B * 32 * VOL2, 256), 256, 0, stream>>>(c2B, gate2, 32, 12, HW2, (size_t)B * 32 * VOL2);

  // ---- c2u ; a0a (in place on c2uB) ----
  convT3d_wmma<32, 16><<<B * 24 * 48 * 10, 32, 0, stream>>>(c2B, pk_c2u, c2u_b, c2u_g, c2u_bb, c2uB, B, 12, 24, 80);
  fuse1x1_3d<16><<<cdiv_i((size_t)B * VOL1, 256), 256, 0, stream>>>(c2uB, c1B, a0a_w, a0a_b, a0a_g, a0a_bb, B, VOL1);

  // ---- a0b ----
  conv3d3_wmma<16, 16, 1, true, false><<<B * 24 * 48 * 10, 32, 0, stream>>>(
      c2uB, pk_a0b, a0b_g, a0b_bb, a0b_b, c1A, B, 24, 48, 160, 24, 48, 160);

  // ---- c1u ; a1a (in place on BUF0) ----
  convT3d_wmma<16, 8><<<B * 48 * 96 * 20, 32, 0, stream>>>(c1A, pk_c1u, c1u_b, c1u_g, c1u_bb, BUF0, B, 24, 48, 160);
  fuse1x1_3d<8><<<cdiv_i((size_t)B * VOL0, 256), 256, 0, stream>>>(BUF0, BUF1, a1a_w, a1a_b, a1a_g, a1a_bb, B, VOL0);

  // ---- a1b ----
  conv3d3_wmma<8, 8, 1, true, false><<<B * 48 * 96 * 20, 32, 0, stream>>>(
      BUF0, pk_a1b, a1b_g, a1b_bb, a1b_b, BUF1, B, 48, 96, 320, 48, 96, 320);

  // ---- oa gate ----
  conv1x1_2d<64, 32, 0><<<cdiv_i((size_t)B * 32 * HW0, 256), 256, 0, stream>>>(featL, oa_w1, oa_b1, oa_g, oa_bb, hid0, B, HW0);
  conv1x1_2d<32, 48, 1><<<cdiv_i((size_t)B * 48 * HW0, 256), 256, 0, stream>>>(hid0, oa_w2, oa_b2, nullptr, nullptr, gate0, B, HW0);
  gate_mul_kernel<<<cdiv_i((size_t)B * 8 * VOL0, 256), 256, 0, stream>>>(BUF1, gate0, 8, 48, HW0, (size_t)B * 8 * VOL0);

  // ---- output conv + softmax ----
  conv3d3_wmma<8, 1, 1, false, true><<<B * 48 * 96 * 20, 32, 0, stream>>>(
      BUF1, pk_out, nullptr, nullptr, nullptr, logits, B, 48, 96, 320, 48, 96, 320);
  softmax48_kernel<<<cdiv_i((size_t)B * HW0, 256), 256, 0, stream>>>(logits, (float*)d_out, B, HW0);
}
